// FlexHeteroAttention_21492016349705
// MI455X (gfx1250) — matrix operand order repaired and checked by reading, not compiled, not run on hardware
//
#include <hip/hip_runtime.h>
#include <hip/hip_bf16.h>

// ---------------- problem constants ----------------
#define N_SRCN   50000
#define N_DSTN   50000
#define N_EDGESN 500000
#define HIDDENN  128
#define HEADSN   8
#define HEAD_DIMN 16
#define SCALEF   0.25f   // HEAD_DIM^-0.5 = 1/4

typedef __attribute__((ext_vector_type(16))) __bf16 v16bf;
typedef __attribute__((ext_vector_type(8)))  float  v8f;

// ---------------- helpers ----------------
__device__ __forceinline__ unsigned short f2bfbits(float f) {
  unsigned int u = __float_as_uint(f);
  u += 0x7FFFu + ((u >> 16) & 1u);     // round-to-nearest-even
  return (unsigned short)(u >> 16);
}
__device__ __forceinline__ __bf16 bits2bf(unsigned short s) {
  return __builtin_bit_cast(__bf16, s);
}
// float atomic max via int-max / uint-min punning (memory must be inited to -inf)
__device__ __forceinline__ void atomicMaxFloat(float* addr, float val) {
  if (val >= 0.0f) atomicMax(reinterpret_cast<int*>(addr), __float_as_int(val));
  else             atomicMin(reinterpret_cast<unsigned int*>(addr), __float_as_uint(val));
}

// ---------------- kernels ----------------

// generic f32 -> bf16 convert
__global__ void k_convert_bf16(const float* __restrict__ in,
                               unsigned short* __restrict__ out, int n) {
  int i = blockIdx.x * blockDim.x + threadIdx.x;
  if (i < n) out[i] = f2bfbits(in[i]);
}

// init m = -inf, s = 0, agg = 0
__global__ void k_init(float* __restrict__ mbuf, float* __restrict__ sbuf,
                       float* __restrict__ agg, int n_ms, int n_agg) {
  int i = blockIdx.x * blockDim.x + threadIdx.x;
  if (i < n_ms) { mbuf[i] = __int_as_float(0xFF800000); sbuf[i] = 0.0f; }
  if (i < n_agg) agg[i] = 0.0f;
}

// Y[M x 128] = Xbf[M x 128] @ Wbf^T + bias   (W is [128 out][128 in], row-major)
// block = 256 threads = 8 waves; block handles 16 rows; wave w handles cols 16w..16w+15
__global__ __launch_bounds__(256) void k_gemm_wmma(
    const unsigned short* __restrict__ Xbf,
    const unsigned short* __restrict__ Wbf,
    const float* __restrict__ bias,
    float* __restrict__ Y, int M) {
  __shared__ unsigned short xs[16 * 128];
  const int tid  = threadIdx.x;
  const int wave = tid >> 5;
  const int lane = tid & 31;
  const int row0 = blockIdx.x * 16;
  if (row0 >= M) return;

  // cooperative load of a 16x128 bf16 activation tile into LDS
  for (int i = tid; i < 16 * 128; i += 256) {
    int r = i >> 7, c = i & 127;
    xs[i] = Xbf[(size_t)(row0 + r) * 128 + c];
  }
  __syncthreads();

  const int n0  = wave * 16;
  const int n   = lane & 15;
  const bool hi = lane >= 16;

  v8f acc = {};
  for (int kt = 0; kt < 4; ++kt) {       // K = 128 = 4 x 32
    const int kb = kt * 32;
    v16bf a, b;
    // A fragment (16x32, lane m = lane&15; lo lanes K {0..7,16..23}, hi lanes +8)
    const int ka = hi ? 8 : 0;
    #pragma unroll
    for (int v = 0; v < 8; ++v) {
      int k = kb + ((v >= 4) ? 16 : 0) + ka + 2 * (v & 3);
      a[2 * v]     = bits2bf(xs[n * 128 + k]);
      a[2 * v + 1] = bits2bf(xs[n * 128 + k + 1]);
    }
    // B fragment (32x16): B[k][n] = W[n0+n][k]; lo lanes K {0..7,16..23}, hi lanes +8
    const int kbB = kb + (hi ? 8 : 0);
    const unsigned short* wrow = Wbf + (size_t)(n0 + n) * 128;
    #pragma unroll
    for (int v = 0; v < 8; ++v) {
      b[2 * v]     = bits2bf(wrow[kbB + ((v >= 4) ? 16 : 0) + 2 * (v & 3)]);
      b[2 * v + 1] = bits2bf(wrow[kbB + ((v >= 4) ? 16 : 0) + 2 * (v & 3) + 1]);
    }
    acc = __builtin_amdgcn_wmma_f32_16x16x32_bf16(
        /*neg_a=*/false, a, /*neg_b=*/false, b,
        /*c_mod=*/(short)0, acc, /*reuse_a=*/false, /*reuse_b=*/false);
  }

  // C layout: VGPR r -> row r (lanes 0-15) / row 8+r (lanes 16-31), col = lane&15
  const float bv   = bias[n0 + n];
  const int   mrow = hi ? 8 : 0;
  #pragma unroll
  for (int r = 0; r < 8; ++r) {
    Y[(size_t)(row0 + mrow + r) * 128 + n0 + n] = acc[r] + bv;
  }
}

// per-edge attention scores + segment max.  one wave per edge, 4 lanes per head.
__global__ __launch_bounds__(256) void k_scores(
    const float* __restrict__ q, const float* __restrict__ kf,
    const long long* __restrict__ ei, const float* __restrict__ edge_bias,
    float* __restrict__ scores, float* __restrict__ mbuf) {
  const int wib  = threadIdx.x >> 5;
  const int lane = threadIdx.x & 31;
  const long long e = (long long)blockIdx.x * 8 + wib;
  if (e >= N_EDGESN) return;
  const int src = (int)ei[e];
  const int dst = (int)ei[N_EDGESN + e];
  const int h = lane >> 2;
  const int c = (lane & 3) * 4;
  const float4 q4 = *reinterpret_cast<const float4*>(q  + (size_t)dst * 128 + h * 16 + c);
  const float4 k4 = *reinterpret_cast<const float4*>(kf + (size_t)src * 128 + h * 16 + c);
  float d = q4.x * k4.x + q4.y * k4.y + q4.z * k4.z + q4.w * k4.w;
  d += __shfl_xor(d, 1, 32);
  d += __shfl_xor(d, 2, 32);
  if ((lane & 3) == 0) {
    float sc = d * SCALEF + edge_bias[h];
    scores[e * 8 + h] = sc;
    atomicMaxFloat(mbuf + (size_t)dst * 8 + h, sc);
  }
}

// exp(score - max) + segment sum.  one thread per (edge, head)
__global__ void k_expsum(float* __restrict__ scores, const long long* __restrict__ ei,
                         const float* __restrict__ mbuf, float* __restrict__ sbuf) {
  long long t = (long long)blockIdx.x * blockDim.x + threadIdx.x;
  if (t >= (long long)N_EDGESN * 8) return;
  long long e = t >> 3;
  int h = (int)(t & 7);
  int dst = (int)ei[N_EDGESN + e];
  float ex = __expf(scores[t] - mbuf[(size_t)dst * 8 + h]);
  scores[t] = ex;
  atomicAdd(sbuf + (size_t)dst * 8 + h, ex);
}

// normalize + scatter-add weighted v.  one wave per edge, 4 lanes per head.
__global__ __launch_bounds__(256) void k_scatter(
    const float* __restrict__ scores, const float* __restrict__ sbuf,
    const float* __restrict__ vf, const long long* __restrict__ ei,
    float* __restrict__ agg) {
  const int wib  = threadIdx.x >> 5;
  const int lane = threadIdx.x & 31;
  const long long e = (long long)blockIdx.x * 8 + wib;
  if (e >= N_EDGESN) return;
  const int src = (int)ei[e];
  const int dst = (int)ei[N_EDGESN + e];
  const int h = lane >> 2;
  const int c = (lane & 3) * 4;
  const float w = scores[e * 8 + h] / (sbuf[(size_t)dst * 8 + h] + 1e-8f);
  const float4 v4 = *reinterpret_cast<const float4*>(vf + (size_t)src * 128 + h * 16 + c);
  float* o = agg + (size_t)dst * 128 + h * 16 + c;
  atomicAdd(o + 0, w * v4.x);
  atomicAdd(o + 1, w * v4.y);
  atomicAdd(o + 2, w * v4.z);
  atomicAdd(o + 3, w * v4.w);
}

// ---------------- launcher ----------------
extern "C" void kernel_launch(void* const* d_in, const int* in_sizes, int n_in,
                              void* d_out, int out_size, void* d_ws, size_t ws_size,
                              hipStream_t stream) {
  (void)in_sizes; (void)n_in; (void)out_size; (void)ws_size;
  const float*     x_src     = (const float*)d_in[0];
  const float*     x_dst     = (const float*)d_in[1];
  const long long* ei        = (const long long*)d_in[2];
  const float*     Wq        = (const float*)d_in[3];
  const float*     bq        = (const float*)d_in[4];
  const float*     Wk        = (const float*)d_in[5];
  const float*     bk        = (const float*)d_in[6];
  const float*     Wv        = (const float*)d_in[7];
  const float*     bv        = (const float*)d_in[8];
  const float*     Wo        = (const float*)d_in[9];
  const float*     bo        = (const float*)d_in[10];
  const float*     edge_bias = (const float*)d_in[11];
  float*           out       = (float*)d_out;

  char* ws = (char*)d_ws;
  size_t off = 0;
  auto alloc = [&](size_t bytes) -> void* {
    void* p = ws + off;
    off = (off + bytes + 255) & ~(size_t)255;
    return p;
  };
  const size_t NFEAT = (size_t)N_DSTN * HIDDENN;       // 6,400,000
  float*          qf       = (float*)alloc(NFEAT * 4);
  float*          kf       = (float*)alloc(NFEAT * 4);
  float*          vf       = (float*)alloc(NFEAT * 4);
  float*          agg      = (float*)alloc(NFEAT * 4);
  float*          scores   = (float*)alloc((size_t)N_EDGESN * HEADSN * 4);
  float*          mbuf     = (float*)alloc((size_t)N_DSTN * HEADSN * 4);
  float*          sbuf     = (float*)alloc((size_t)N_DSTN * HEADSN * 4);
  unsigned short* xsrc_bf  = (unsigned short*)alloc(NFEAT * 2);
  unsigned short* xdst_bf  = (unsigned short*)alloc(NFEAT * 2);
  unsigned short* agg_bf   = (unsigned short*)alloc(NFEAT * 2);
  unsigned short* wq_bf    = (unsigned short*)alloc(HIDDENN * HIDDENN * 2);
  unsigned short* wk_bf    = (unsigned short*)alloc(HIDDENN * HIDDENN * 2);
  unsigned short* wv_bf    = (unsigned short*)alloc(HIDDENN * HIDDENN * 2);
  unsigned short* wo_bf    = (unsigned short*)alloc(HIDDENN * HIDDENN * 2);

  const int nfeat_i = (int)NFEAT;                       // 6,400,000
  const int nW      = HIDDENN * HIDDENN;                // 16,384
  const int cvt_blk = 256;

  // 1) convert inputs to bf16
  k_convert_bf16<<<(nfeat_i + cvt_blk - 1) / cvt_blk, cvt_blk, 0, stream>>>(x_src, xsrc_bf, nfeat_i);
  k_convert_bf16<<<(nfeat_i + cvt_blk - 1) / cvt_blk, cvt_blk, 0, stream>>>(x_dst, xdst_bf, nfeat_i);
  k_convert_bf16<<<(nW + cvt_blk - 1) / cvt_blk, cvt_blk, 0, stream>>>(Wq, wq_bf, nW);
  k_convert_bf16<<<(nW + cvt_blk - 1) / cvt_blk, cvt_blk, 0, stream>>>(Wk, wk_bf, nW);
  k_convert_bf16<<<(nW + cvt_blk - 1) / cvt_blk, cvt_blk, 0, stream>>>(Wv, wv_bf, nW);
  k_convert_bf16<<<(nW + cvt_blk - 1) / cvt_blk, cvt_blk, 0, stream>>>(Wo, wo_bf, nW);

  // 2) init segment max / sum / aggregate buffers
  k_init<<<(nfeat_i + cvt_blk - 1) / cvt_blk, cvt_blk, 0, stream>>>(
      mbuf, sbuf, agg, N_DSTN * HEADSN, nfeat_i);

  // 3) Q/K/V projections via bf16 WMMA (50000 rows = 3125 tiles of 16)
  const int gemm_grid = N_DSTN / 16;  // 3125
  k_gemm_wmma<<<gemm_grid, 256, 0, stream>>>(xdst_bf, wq_bf, bq, qf, N_DSTN);
  k_gemm_wmma<<<gemm_grid, 256, 0, stream>>>(xsrc_bf, wk_bf, bk, kf, N_SRCN);
  k_gemm_wmma<<<gemm_grid, 256, 0, stream>>>(xsrc_bf, wv_bf, bv, vf, N_SRCN);

  // 4) edge phase: scores+max, exp+sum, normalize+scatter
  const int edge_grid = N_EDGESN / 8;  // 62500 blocks, 8 waves (edges) each
  k_scores<<<edge_grid, 256, 0, stream>>>(qf, kf, ei, edge_bias, scores, mbuf);
  const long long nEH = (long long)N_EDGESN * HEADSN;   // 4,000,000
  k_expsum<<<(int)((nEH + 255) / 256), 256, 0, stream>>>(scores, ei, mbuf, sbuf);
  k_scatter<<<edge_grid, 256, 0, stream>>>(scores, sbuf, vf, ei, agg);

  // 5) output projection
  k_convert_bf16<<<(nfeat_i + cvt_blk - 1) / cvt_blk, cvt_blk, 0, stream>>>(agg, agg_bf, nfeat_i);
  k_gemm_wmma<<<gemm_grid, 256, 0, stream>>>(agg_bf, wo_bf, bo, out, N_DSTN);
}
